// DeformableConv2d_89593017795328
// MI455X (gfx1250) — matrix operand order repaired and checked by reading, not compiled, not run on hardware
//
#include <hip/hip_runtime.h>

// ---------------------------------------------------------------------------
// Deformable conv pipeline for MI455X (gfx1250, wave32, WMMA 16x16x32 f16)
// B=4, C=64, H=W=256, DG=8, K=9 taps. All convs as implicit GEMM on WMMA.
// Activations in NHWC f16 (33.5MB/layer -> resident in 192MB L2).
// Round 2: 32x32 wave tiles (2x2 WMMA blocking, 16 FLOP/byte) + zero-page
// boundary handling (no exec divergence in the K-loop).
// ---------------------------------------------------------------------------

typedef __attribute__((ext_vector_type(16))) _Float16 v16h;
typedef __attribute__((ext_vector_type(8)))  _Float16 v8h;
typedef __attribute__((ext_vector_type(8)))  float    v8f;

union V16 { v16h v; v8h h[2]; };

#define BATCH 4
#define CCH   64
#define HH    256
#define WW    256
#define HWSZ  65536
#define DGRP  8

// ---------------------------------------------------------------------------
__global__ void zero_pad_kernel(_Float16* __restrict__ p) {
  p[blockIdx.x * 256 + threadIdx.x] = (_Float16)0.f;
}

// Weight repack: NCHW-f32 (Co,Cin,3,3) -> f16 [coP][t][cin]  (t = ki*3+kj)
// dcnMode: -> f16 [t][coP][cin] (per-tap GEMM A matrices)
__global__ void pack_w_kernel(const float* __restrict__ w, _Float16* __restrict__ dst,
                              int Cin, int Co, int CoP, int dcnMode) {
  int n = CoP * 9 * Cin;
  for (int i = blockIdx.x * blockDim.x + threadIdx.x; i < n;
       i += gridDim.x * blockDim.x) {
    int ci = i % Cin;
    int t  = (i / Cin) % 9;
    int co = i / (Cin * 9);
    float v = (co < Co) ? w[((size_t)co * Cin + ci) * 9 + t] : 0.f;
    size_t di = dcnMode ? (((size_t)t * CoP + co) * Cin + ci)
                        : (((size_t)co * 9 + t) * Cin + ci);
    dst[di] = (_Float16)v;
  }
}

// ---------------------------------------------------------------------------
// Prep: flow-warp supp (bilinear, zero pad), feat0 = concat(warped, ref) NHWC
// f16 (128ch), suppH = supp NHWC f16. One thread per pixel.
// ---------------------------------------------------------------------------
__global__ __launch_bounds__(256) void prep_kernel(
    const float* __restrict__ ref, const float* __restrict__ supp,
    const float* __restrict__ flow, _Float16* __restrict__ feat0,
    _Float16* __restrict__ suppH) {
  int idx = blockIdx.x * 256 + threadIdx.x;          // 0 .. B*H*W-1
  int x = idx & 255;
  int y = (idx >> 8) & 255;
  int b = idx >> 16;
  size_t pix = (size_t)idx;                          // == (b*H+y)*W + x
  float fx = flow[((size_t)b * 2 + 0) * HWSZ + (y << 8) + x];
  float fy = flow[((size_t)b * 2 + 1) * HWSZ + (y << 8) + x];
  float sy = (float)y + fy, sx = (float)x + fx;
  float y0f = floorf(sy), x0f = floorf(sx);
  float wy = sy - y0f, wx = sx - x0f;
  int y0 = (int)y0f, x0 = (int)x0f;
  bool ry0 = (y0 >= 0) && (y0 < HH), ry1 = (y0 + 1 >= 0) && (y0 + 1 < HH);
  bool rx0 = (x0 >= 0) && (x0 < WW), rx1 = (x0 + 1 >= 0) && (x0 + 1 < WW);
  float w00 = (1.f - wy) * (1.f - wx), w01 = (1.f - wy) * wx;
  float w10 = wy * (1.f - wx),         w11 = wy * wx;
  const float* sb = supp + (size_t)b * CCH * HWSZ;
  for (int c = 0; c < CCH; ++c) {
    const float* sc = sb + (size_t)c * HWSZ;
    float a00 = (ry0 && rx0) ? sc[y0 * WW + x0]           : 0.f;
    float a01 = (ry0 && rx1) ? sc[y0 * WW + x0 + 1]       : 0.f;
    float a10 = (ry1 && rx0) ? sc[(y0 + 1) * WW + x0]     : 0.f;
    float a11 = (ry1 && rx1) ? sc[(y0 + 1) * WW + x0 + 1] : 0.f;
    float val = a00 * w00 + a01 * w01 + a10 * w10 + a11 * w11;
    feat0[pix * 128 + c]      = (_Float16)val;
    feat0[pix * 128 + 64 + c] = (_Float16)ref[((size_t)b * CCH + c) * HWSZ + (y << 8) + x];
    suppH[pix * 64 + c]       = (_Float16)sc[(y << 8) + x];
  }
}

// ---------------------------------------------------------------------------
// Implicit-GEMM 3x3 conv, WMMA f32<-f16. One wave = 32(cout) x 32(px) tile:
// 2x2 blocking of 16x16 WMMA tiles -> each A/B fragment reused twice,
// 4 back-to-back wmma per K-step. K = 9 taps x Cin (chunks of 32).
// OOB columns read from a zeroed pad page (per-lane address cndmask, no exec
// divergence). act: 0=none 1=relu 2=leaky(0.1). out16: NHWC f16 stride CoP.
// acc32: NHWC f32 64ch residual accumulator; accMode 0=off 1=init 2=add*scale.
// ---------------------------------------------------------------------------
__global__ __launch_bounds__(256) void conv3x3_wmma_kernel(
    const _Float16* __restrict__ in, int Cin,
    const _Float16* __restrict__ wq, const float* __restrict__ bias,
    int CoP, int CoValid, int dil, int act,
    _Float16* __restrict__ out16, float* __restrict__ acc32,
    float accScale, int accMode, const _Float16* __restrict__ zeroPad) {
  int wid  = (blockIdx.x << 3) + (threadIdx.x >> 5);
  int lane = threadIdx.x & 31;
  int nCoT = CoP >> 5;                     // 32-wide cout tiles
  int coT  = wid % nCoT;
  int r    = wid / nCoT;
  int xt   = r & 7;                        // 8 x-tiles of 32 px
  int y    = (r >> 3) & 255;
  int b    = r >> 11;
  int coBase = coT << 5, xBase = xt << 5;
  int mrow = lane & 15;
  int hiH  = (lane >> 4) & 1;
  int kOffA = hiH * 8;                     // A lanes 16-31: K {8..15, 24..31}
  int kOffB = hiH * 8;                     // B lanes 16-31: K {8..15, 24..31}
  const _Float16* aRow0 = wq + ((size_t)(coBase + mrow)) * 9 * Cin + kOffA;
  const _Float16* aRow1 = aRow0 + (size_t)16 * 9 * Cin;
  int xPix0 = xBase + mrow, xPix1 = xPix0 + 16;
  const _Float16* zp = zeroPad + kOffB;
  v8f c00 = {}, c01 = {}, c10 = {}, c11 = {};
  int KC = Cin >> 5;
  for (int t = 0; t < 9; ++t) {
    int yy = y + dil * ((t / 3) - 1);
    if (yy < 0 || yy >= HH) continue;      // uniform per wave
    int dx = dil * ((t % 3) - 1);
    int xx0 = xPix0 + dx, xx1 = xPix1 + dx;
    const _Float16* rowPtr = in + ((size_t)b * HH + yy) * WW * Cin + kOffB;
    const _Float16* bP0 = (xx0 >= 0 && xx0 < WW) ? rowPtr + (size_t)xx0 * Cin : zp;
    const _Float16* bP1 = (xx1 >= 0 && xx1 < WW) ? rowPtr + (size_t)xx1 * Cin : zp;
    const _Float16* aP0 = aRow0 + t * Cin;
    const _Float16* aP1 = aRow1 + t * Cin;
    for (int kc = 0; kc < KC; ++kc) {
      V16 a0, a1, b0, b1;
      const v8h* p;
      p = (const v8h*)(aP0 + kc * 32); a0.h[0] = p[0]; a0.h[1] = p[2];
      p = (const v8h*)(aP1 + kc * 32); a1.h[0] = p[0]; a1.h[1] = p[2];
      p = (const v8h*)(bP0 + kc * 32); b0.h[0] = p[0]; b0.h[1] = p[2];
      p = (const v8h*)(bP1 + kc * 32); b1.h[0] = p[0]; b1.h[1] = p[2];
      c00 = __builtin_amdgcn_wmma_f32_16x16x32_f16(false, a0.v, false, b0.v,
                                                   (short)0, c00, false, false);
      c01 = __builtin_amdgcn_wmma_f32_16x16x32_f16(false, a0.v, false, b1.v,
                                                   (short)0, c01, false, false);
      c10 = __builtin_amdgcn_wmma_f32_16x16x32_f16(false, a1.v, false, b0.v,
                                                   (short)0, c10, false, false);
      c11 = __builtin_amdgcn_wmma_f32_16x16x32_f16(false, a1.v, false, b1.v,
                                                   (short)0, c11, false, false);
    }
  }
  // Epilogue: D element (M=j+hiH*8, N=mrow) per 16x16 sub-tile.
  size_t pixA = ((size_t)b * HH + y) * WW + xBase + mrow;
  size_t pixB = pixA + 16;
  auto emit = [&](v8f cc, int co0, size_t pix) {
    float vals[8];
#pragma unroll
    for (int j = 0; j < 8; ++j) {
      int co = co0 + j;
      float v = cc[j] + ((co < CoValid) ? bias[co] : 0.f);
      if (act == 1)      v = v > 0.f ? v : 0.f;
      else if (act == 2) v = v > 0.f ? v : 0.1f * v;
      vals[j] = v;
    }
    if (out16) {
      v8h hv;
#pragma unroll
      for (int j = 0; j < 8; ++j) hv[j] = (_Float16)vals[j];
      *(v8h*)(out16 + pix * CoP + co0) = hv;          // 16B aligned
    }
    if (accMode) {
      float* ap2 = acc32 + pix * 64 + co0;
#pragma unroll
      for (int j = 0; j < 8; ++j)
        ap2[j] = (accMode == 1) ? vals[j] : ap2[j] + accScale * vals[j];
    }
  };
  int co0 = coBase + hiH * 8;
  emit(c00, co0,      pixA);
  emit(c01, co0,      pixB);
  emit(c10, co0 + 16, pixA);
  emit(c11, co0 + 16, pixB);
}

// acc f32 -> feat_final f16 (elementwise)
__global__ void cvt_kernel(const float* __restrict__ src,
                           _Float16* __restrict__ dst, int n) {
  for (int i = blockIdx.x * blockDim.x + threadIdx.x; i < n;
       i += gridDim.x * blockDim.x)
    dst[i] = (_Float16)src[i];
}

// ---------------------------------------------------------------------------
// Deformable stage: per block = one (b, y, 16-pixel x-tile).
// Phase 1 (128 thr): sample supp bilinearly with offsets=25*tanh(raw)+flow,
//   modulate by sigmoid(mask), stage 9 B-tiles [16px][64cin] into LDS.
// Phase 2 (4 waves): per-tap 64x64 GEMM via WMMA; out += dcn_w[:,:,t] @ s_t.
// ---------------------------------------------------------------------------
__global__ __launch_bounds__(128) void deform_wmma_kernel(
    const _Float16* __restrict__ suppH, const _Float16* __restrict__ offRaw,
    const float* __restrict__ flow, const _Float16* __restrict__ wdcn,
    const float* __restrict__ dcnB, float* __restrict__ out) {
  __shared__ _Float16 Bs[9][16][80];                  // pad->160B rows, 16B aligned
  int xt = blockIdx.x & 15;
  int y  = (blockIdx.x >> 4) & 255;
  int b  = blockIdx.x >> 12;
  int xBase = xt << 4;
  int tid = threadIdx.x;

  for (int task = tid; task < 1152; task += 128) {    // 9 taps * 16 px * 8 grp
    int g  = task & 7;
    int pn = (task >> 3) & 15;
    int t  = task >> 7;
    int x  = xBase + pn;
    size_t pix = ((size_t)b * HH + y) * WW + x;
    float fx = flow[((size_t)b * 2 + 0) * HWSZ + (y << 8) + x];
    float fy = flow[((size_t)b * 2 + 1) * HWSZ + (y << 8) + x];
    const _Float16* ob = offRaw + pix * 224;          // padded 224-ch NHWC
    int cdy = g * 18 + t * 2;
    float dyv = 25.f * tanhf((float)ob[cdy])     + ((cdy     < 72) ? fx : fy);
    float dxv = 25.f * tanhf((float)ob[cdy + 1]) + ((cdy + 1 < 72) ? fx : fy);
    int cm = 144 + g * 9 + t;
    float mk = 1.f / (1.f + expf(-(float)ob[cm]));
    float syf = (float)y + (float)(t / 3 - 1) + dyv;
    float sxf = (float)x + (float)(t % 3 - 1) + dxv;
    float y0f = floorf(syf), x0f = floorf(sxf);
    float wy = syf - y0f, wx = sxf - x0f;
    int y0 = (int)y0f, x0 = (int)x0f;
    float acc8[8];
#pragma unroll
    for (int k = 0; k < 8; ++k) acc8[k] = 0.f;
#pragma unroll
    for (int cy = 0; cy < 2; ++cy) {
#pragma unroll
      for (int cx = 0; cx < 2; ++cx) {
        int yi = y0 + cy, xi = x0 + cx;
        if (yi >= 0 && yi < HH && xi >= 0 && xi < WW) {
          float wgt = (cy ? wy : 1.f - wy) * (cx ? wx : 1.f - wx);
          const v8h* sp = (const v8h*)(suppH +
              (((size_t)b * HH + yi) * WW + xi) * 64 + g * 8);
          v8h sv = *sp;
#pragma unroll
          for (int k = 0; k < 8; ++k) acc8[k] += wgt * (float)sv[k];
        }
      }
    }
    v8h hv;
#pragma unroll
    for (int k = 0; k < 8; ++k) hv[k] = (_Float16)(acc8[k] * mk);
    *(v8h*)&Bs[t][pn][g * 8] = hv;
  }
  __syncthreads();

  int wv = tid >> 5, lane = tid & 31;
  int mrow = lane & 15, hiH = (lane >> 4) & 1;
  int coBase = wv << 4;
  int co0 = coBase + hiH * 8;
  v8f c = {};
#pragma unroll
  for (int j = 0; j < 8; ++j) c[j] = dcnB[co0 + j];
  for (int t = 0; t < 9; ++t) {
#pragma unroll
    for (int kc = 0; kc < 2; ++kc) {
      V16 a, bb;
      const v8h* ap = (const v8h*)(wdcn +
          ((size_t)(t * 64 + coBase + mrow)) * 64 + kc * 32 + hiH * 8);
      a.h[0] = ap[0];
      a.h[1] = ap[2];
      const v8h* bp = (const v8h*)(&Bs[t][mrow][kc * 32 + hiH * 8]);
      bb.h[0] = bp[0];
      bb.h[1] = bp[2];
      c = __builtin_amdgcn_wmma_f32_16x16x32_f16(false, a.v, false, bb.v,
                                                 (short)0, c, false, false);
    }
  }
  int xo = xBase + mrow;
#pragma unroll
  for (int j = 0; j < 8; ++j) {
    int co = co0 + j;
    out[(((size_t)b * 64 + co) * HH + y) * WW + xo] = c[j];
  }
}

// ---------------------------------------------------------------------------
extern "C" void kernel_launch(void* const* d_in, const int* in_sizes, int n_in,
                              void* d_out, int out_size, void* d_ws, size_t ws_size,
                              hipStream_t stream) {
  const float* ref   = (const float*)d_in[0];
  const float* supp  = (const float*)d_in[1];
  const float* flow  = (const float*)d_in[2];
  const float* fc_w  = (const float*)d_in[3];
  const float* fc_b  = (const float*)d_in[4];
  const float* cw[6]; const float* cb[6];
  for (int i = 0; i < 6; ++i) {
    cw[i] = (const float*)d_in[5 + 2 * i];
    cb[i] = (const float*)d_in[6 + 2 * i];
  }
  const float* off_w = (const float*)d_in[17];
  const float* off_b = (const float*)d_in[18];
  const float* dcn_w = (const float*)d_in[19];
  const float* dcn_b = (const float*)d_in[20];

  char* ws = (char*)d_ws;
  size_t o = 0;
  auto take = [&](size_t bytes) {
    size_t r = o;
    o += (bytes + 255) & ~(size_t)255;
    return r;
  };
  _Float16* zeroPad = (_Float16*)(ws + take(1024));
  _Float16* wqFc  = (_Float16*)(ws + take((size_t)64 * 9 * 128 * 2));
  _Float16* wqC[6];
  for (int i = 0; i < 6; ++i) wqC[i] = (_Float16*)(ws + take((size_t)64 * 9 * 64 * 2));
  _Float16* wqOff = (_Float16*)(ws + take((size_t)224 * 9 * 64 * 2));
  _Float16* wqDcn = (_Float16*)(ws + take((size_t)9 * 64 * 64 * 2));
  _Float16* suppH = (_Float16*)(ws + take((size_t)BATCH * HWSZ * 64 * 2));
  _Float16* featA = (_Float16*)(ws + take((size_t)BATCH * HWSZ * 64 * 2));
  _Float16* tmp   = (_Float16*)(ws + take((size_t)BATCH * HWSZ * 64 * 2));
  _Float16* feat0 = (_Float16*)(ws + take((size_t)BATCH * HWSZ * 128 * 2));
  float*    acc   = (float*)   (ws + take((size_t)BATCH * HWSZ * 64 * 4));
  // off conv output (224ch f16, 117MB) aliases feat0+acc (134MB), both dead then
  _Float16* offOut = feat0;
  float* out = (float*)d_out;

  // 1) zero pad page + pack weights
  zero_pad_kernel<<<2, 256, 0, stream>>>(zeroPad);
  pack_w_kernel<<<256, 256, 0, stream>>>(fc_w, wqFc, 128, 64, 64, 0);
  for (int i = 0; i < 6; ++i)
    pack_w_kernel<<<256, 256, 0, stream>>>(cw[i], wqC[i], 64, 64, 64, 0);
  pack_w_kernel<<<256, 256, 0, stream>>>(off_w, wqOff, 64, 216, 224, 0);
  pack_w_kernel<<<256, 256, 0, stream>>>(dcn_w, wqDcn, 64, 64, 64, 1);

  // 2) flow-warp + NHWC f16 packing
  prep_kernel<<<1024, 256, 0, stream>>>(ref, supp, flow, feat0, suppH);

  const int convBlocks64  = (BATCH * HH * 8 * 2) / 8;     // 32x32 tiles, CoP=64
  const int convBlocks224 = (BATCH * HH * 8 * 7) / 8;     // CoP=224

  // 3) fc conv (2C->C), leaky 0.1; featA = result; acc := featA
  conv3x3_wmma_kernel<<<convBlocks64, 256, 0, stream>>>(
      feat0, 128, wqFc, fc_b, 64, 64, 1, 2, featA, acc, 1.f, 1, zeroPad);
  // 4) branch 1: acc += relu(conv(relu(conv(featA,c1)),c2))
  conv3x3_wmma_kernel<<<convBlocks64, 256, 0, stream>>>(
      featA, 64, wqC[0], cb[0], 64, 64, 1, 1, tmp, (float*)0, 0.f, 0, zeroPad);
  conv3x3_wmma_kernel<<<convBlocks64, 256, 0, stream>>>(
      tmp, 64, wqC[1], cb[1], 64, 64, 1, 1, (_Float16*)0, acc, 1.f, 2, zeroPad);
  // 5) branch 2 (dil 2,2): acc += 0.1*b2
  conv3x3_wmma_kernel<<<convBlocks64, 256, 0, stream>>>(
      featA, 64, wqC[2], cb[2], 64, 64, 2, 1, tmp, (float*)0, 0.f, 0, zeroPad);
  conv3x3_wmma_kernel<<<convBlocks64, 256, 0, stream>>>(
      tmp, 64, wqC[3], cb[3], 64, 64, 2, 1, (_Float16*)0, acc, 0.1f, 2, zeroPad);
  // 6) branch 3 (dil 2 then 4): acc += 0.1*b3
  conv3x3_wmma_kernel<<<convBlocks64, 256, 0, stream>>>(
      featA, 64, wqC[4], cb[4], 64, 64, 2, 1, tmp, (float*)0, 0.f, 0, zeroPad);
  conv3x3_wmma_kernel<<<convBlocks64, 256, 0, stream>>>(
      tmp, 64, wqC[5], cb[5], 64, 64, 4, 1, (_Float16*)0, acc, 0.1f, 2, zeroPad);
  // 7) feat_final (f16) <- acc ; reuse featA region
  cvt_kernel<<<4096, 256, 0, stream>>>(acc, featA, BATCH * HWSZ * 64);
  // 8) offset conv (C->216, padded 224), no activation
  conv3x3_wmma_kernel<<<convBlocks224, 256, 0, stream>>>(
      featA, 64, wqOff, off_b, 224, 216, 1, 0, offOut, (float*)0, 0.f, 0, zeroPad);
  // 9) deformable sampling + per-tap 64x64 WMMA GEMMs -> d_out (NCHW f32)
  deform_wmma_kernel<<<BATCH * HH * 16, 128, 0, stream>>>(
      suppH, offOut, flow, wqDcn, dcn_b, out);
}
